// audio2poseLSTM_18073222381799
// MI455X (gfx1250) — matrix operand order repaired and checked by reading, not compiled, not run on hardware
//
#include <hip/hip_runtime.h>
#include <hip/hip_bf16.h>

// ---------------------------------------------------------------------------
// audio2pose (ResNet34 x2 + 2-layer LSTM) for MI455X / gfx1250.
//
// Matrix engine: v_wmma_f32_16x16x32_bf16, wave32, fp32 accumulate.
//  - conv: implicit GEMM; weights pre-transposed to [OC][Kpad] (K-major,
//    zero-padded to 32) so BOTH A and B fragments are pairs of aligned 16B
//    global_load_b128 (fragment K-contiguity, ISA 7.12.2). No LDS, no
//    barriers, no guards in hot loops. 8 waves/block share a 64-col strip
//    (cache reuse); wave = 16x64 tile = 4 WMMA accumulators.
//  - dense GEMM: B in natural [N,K] layout (PyTorch [out,in]) -> same
//    two-b128 fragment loads; FC/LSTM weights only need f32->bf16 cast.
//  - BN/ReLU/residual fused in conv epilogue, dual f32+bf16 outputs.
// ---------------------------------------------------------------------------

typedef __bf16 bf16;
typedef __attribute__((ext_vector_type(16))) __bf16 v16bf;
typedef __attribute__((ext_vector_type(8)))  __bf16 v8bf;
typedef __attribute__((ext_vector_type(8)))  float  v8f;

// CDNA5 ISA 7.12.2 fragment layouts (16-bit A 16x32 / B 32x16 / f32 C 16x16):
//   A: lane l row M=l&15; a[0..7]  = K of  (l>=16?8:0) + 0..7
//                         a[8..15] = K of 16+(l>=16?8:0) + 0..7   [two 16B runs]
//   B: lane l col N=l&15; b[] K order as a[] above; this code uses a[i], b[i] = K of (l>=16?16:0) + i
//   C/D: reg r, lane l: row M = r + (l>=16)*8, col N = l&15
__device__ __forceinline__ v8f wmma_bf16(v16bf a, v16bf b, v8f c) {
  return __builtin_amdgcn_wmma_f32_16x16x32_bf16(false, a, false, b,
                                                 (short)0, c, false, false);
}

__device__ __forceinline__ v16bf frag_from2(v8bf l0, v8bf l1) {
  v16bf f;
#pragma unroll
  for (int j = 0; j < 8; ++j) { f[j] = l0[j]; f[8 + j] = l1[j]; }
  return f;
}

// ---------------------------------------------------------------------------
// Implicit-GEMM conv. Block = 256 thr = 8 waves -> 128(M) x 64(OC) tile.
// x NHWC bf16; wt[OC][Kpad] bf16 (K = (kh*KW+kw)*C + c, zero-padded);
// OC % 64 == 0, Kpad % 32 == 0.
// ---------------------------------------------------------------------------
template <bool CFAST>
__global__ void conv_wmma_kernel(
    const bf16* __restrict__ x, const bf16* __restrict__ wt,
    const float* __restrict__ gamma, const float* __restrict__ beta,
    const float* __restrict__ res,
    float* __restrict__ yf, bf16* __restrict__ yb,
    int N, int H, int W, int C, int OC, int OH, int OW,
    int KH, int KW, int stride, int pad, int relu, int Kpad)
{
  const int wave  = threadIdx.x >> 5;
  const int lane  = threadIdx.x & 31;
  const int khalf = lane >> 4;
  const int colL  = lane & 15;
  const long M    = (long)N * OH * OW;
  const int  nbn  = OC >> 6;
  const long tmblk = (long)blockIdx.x / nbn;
  const int  tnblk = (int)((long)blockIdx.x % nbn);
  const long row0  = (tmblk * 8 + wave) * 16;
  const int  col0  = tnblk * 64;
  if (row0 >= M) return;   // no barriers in this kernel

  // A-side row decode (per lane)
  const long arow = row0 + colL;
  const bool arow_ok = arow < M;
  int n = 0, oh = 0, ow = 0;
  if (arow_ok) {
    n = (int)(arow / ((long)OH * OW));
    int r = (int)(arow % ((long)OH * OW));
    oh = r / OW; ow = r % OW;
  }
  const int ih0 = oh * stride - pad;
  const int iw0 = ow * stride - pad;

  // per-lane weight row (K-major); each lane streams sequentially through it
  const bf16* bp0 = wt + (long)(col0 + colL) * Kpad + (khalf << 4);
  const long  rstride = (long)16 * Kpad;   // 16 cols between sub-tiles

  v8f z = {0.f, 0.f, 0.f, 0.f, 0.f, 0.f, 0.f, 0.f};
  v8f acc[4] = {z, z, z, z};
  const int Ktot = KH * KW * C;

  if (CFAST) {
    // C % 32 == 0: no div/mod, no guards in the hot loop.
    for (int kh = 0; kh < KH; ++kh) {
      const int ih = ih0 + kh;
      for (int kw = 0; kw < KW; ++kw) {
        const int iw = iw0 + kw;
        const bool pv = arow_ok && ih >= 0 && ih < H && iw >= 0 && iw < W;
        const bf16* xp = x + (((long)n * H + ih) * W + iw) * C;
        const int kbase = (kh * KW + kw) * C;
        for (int c0 = 0; c0 < C; c0 += 32) {
          const int k0 = kbase + c0;
          __builtin_prefetch(bp0 + k0 + 64, 0, 1);   // speculative stream-ahead
          // A fragment: two aligned 16B loads + pointer select for padding
          v16bf a;
          {
            const bf16* pa = (pv ? xp + c0 : x) + (khalf << 4);
            v8bf r0 = *(const v8bf*)pa;
            v8bf r1 = *(const v8bf*)(pa + 8);
#pragma unroll
            for (int j = 0; j < 8; ++j) {
              a[j]     = pv ? r0[j] : (bf16)0.0f;
              a[8 + j] = pv ? r1[j] : (bf16)0.0f;
            }
          }
          // B fragments: batch all 8 b128 loads, then 4 WMMAs
          v16bf bf[4];
#pragma unroll
          for (int i = 0; i < 4; ++i) {
            const bf16* bp = bp0 + i * rstride + k0;
            bf[i] = frag_from2(*(const v8bf*)bp, *(const v8bf*)(bp + 8));
          }
#pragma unroll
          for (int i = 0; i < 4; ++i) acc[i] = wmma_bf16(a, bf[i], acc[i]);
        }
      }
    }
  } else {
    // generic path (stem convs only: C=1 or 3; Kpad <= 160 -> <= 5 chunks)
    for (int k0 = 0; k0 < Kpad; k0 += 32) {
      v16bf a;
#pragma unroll
      for (int e = 0; e < 16; ++e) {
        int k = k0 + (khalf << 4) + e;
        bf16 val = (bf16)0.0f;
        if (arow_ok && k < Ktot) {
          int c  = k % C;
          int kw = (k / C) % KW;
          int kh = k / (C * KW);
          int ih = ih0 + kh, iw = iw0 + kw;
          if (ih >= 0 && ih < H && iw >= 0 && iw < W)
            val = x[(((long)n * H + ih) * W + iw) * C + c];
        }
        a[e] = val;
      }
      v16bf bf[4];
#pragma unroll
      for (int i = 0; i < 4; ++i) {
        const bf16* bp = bp0 + i * rstride + k0;   // pad region is zeros
        bf[i] = frag_from2(*(const v8bf*)bp, *(const v8bf*)(bp + 8));
      }
#pragma unroll
      for (int i = 0; i < 4; ++i) acc[i] = wmma_bf16(a, bf[i], acc[i]);
    }
  }

  // epilogue: BN + residual + ReLU, dual-precision store (row-edge guarded)
#pragma unroll
  for (int i = 0; i < 4; ++i) {
    const int col = col0 + 16 * i + colL;
    const float g  = gamma ? gamma[col] : 1.f;
    const float bb = beta  ? beta[col]  : 0.f;
#pragma unroll
    for (int r = 0; r < 8; ++r) {
      long rowm = row0 + r + (khalf << 3);
      if (rowm >= M) continue;
      long off = rowm * OC + col;
      float v = acc[i][r] * g + bb;
      if (res)  v += res[off];
      if (relu) v = v > 0.f ? v : 0.f;
      if (yf) yf[off] = v;
      if (yb) yb[off] = (bf16)v;
    }
  }
}

// ---------------------------------------------------------------------------
// Dense GEMM: C[M,N] (=|+=) A[M,K] * Bt[N,K]^T (+bias); wave = 16x64 tile.
// Requirements (all call sites satisfy): M%16==0, Nn%64==0, K%32==0.
// Bt is the NATURAL [N,K] row-major layout (PyTorch [out,in] weights).
// ---------------------------------------------------------------------------
__global__ void gemm_wmma_kernel(
    const bf16* __restrict__ A, const bf16* __restrict__ Bt,
    float* __restrict__ C, int M, int Nn, int K,
    const float* __restrict__ bias, int accumulate)
{
  const int wave  = threadIdx.x >> 5;
  const int lane  = threadIdx.x & 31;
  const int khalf = lane >> 4;
  const int colL  = lane & 15;
  const int  ntn = Nn >> 6;
  const long ntm = M >> 4;
  long tile = (long)blockIdx.x * (blockDim.x >> 5) + wave;
  if (tile >= ntm * (long)ntn) return;
  const int  tn = (int)(tile % ntn);
  const long tm = tile / ntn;
  const long row0 = tm * 16;
  const int  col0 = tn * 64;

  const bf16* ap  = A + (row0 + colL) * (long)K + (khalf << 4);
  const bf16* bp0 = Bt + (long)(col0 + colL) * K + (khalf << 4);
  const long  rstride = (long)16 * K;

  v8f z = {0.f, 0.f, 0.f, 0.f, 0.f, 0.f, 0.f, 0.f};
  v8f acc[4] = {z, z, z, z};

  for (int k0 = 0; k0 < K; k0 += 32) {
    __builtin_prefetch(bp0 + k0 + 64, 0, 1);   // speculative
    v16bf a = frag_from2(*(const v8bf*)(ap + k0), *(const v8bf*)(ap + k0 + 8));
    v16bf bf[4];
#pragma unroll
    for (int i = 0; i < 4; ++i) {
      const bf16* bp = bp0 + i * rstride + k0;
      bf[i] = frag_from2(*(const v8bf*)bp, *(const v8bf*)(bp + 8));
    }
#pragma unroll
    for (int i = 0; i < 4; ++i) acc[i] = wmma_bf16(a, bf[i], acc[i]);
  }

#pragma unroll
  for (int i = 0; i < 4; ++i) {
    const int col = col0 + 16 * i + colL;
    const float bb = bias ? bias[col] : 0.f;
#pragma unroll
    for (int r = 0; r < 8; ++r) {
      long rowm = row0 + r + (khalf << 3);
      long off = rowm * Nn + col;
      float v = acc[i][r] + bb;
      if (accumulate) v += C[off];
      C[off] = v;
    }
  }
}

// ---------------------------------------------------------------------------
// Small support kernels
// ---------------------------------------------------------------------------
__global__ void nchw_to_nhwc_bf(const float* __restrict__ x, bf16* __restrict__ y,
                                int N, int C, int H, int W)
{
  long total = (long)N * C * H * W;
  long idx = (long)blockIdx.x * blockDim.x + threadIdx.x;
  if (idx >= total) return;
  int c = (int)(idx % C); long r = idx / C;
  int w = (int)(r % W); r /= W;
  int h = (int)(r % H); int n = (int)(r / H);
  y[idx] = (bf16)x[(((long)n * C + c) * H + h) * W + w];
}

// OIHW f32 -> [O][Kpad] bf16, K = (kh*Kk+kw)*I + i, zero-padded to Kpad
__global__ void conv_w_transform(const float* __restrict__ w, bf16* __restrict__ wt,
                                 int O, int I, int Kk, int Kpad)
{
  long total = (long)O * Kpad;
  long idx = (long)blockIdx.x * blockDim.x + threadIdx.x;
  if (idx >= total) return;
  int k = (int)(idx % Kpad);
  int o = (int)(idx / Kpad);
  bf16 v = (bf16)0.0f;
  if (k < I * Kk * Kk) {
    int i = k % I; int r = k / I;
    int kw = r % Kk; int kh = r / Kk;
    v = (bf16)w[(((long)o * I + i) * Kk + kh) * Kk + kw];
  }
  wt[idx] = v;
}

__global__ void cast_bf(const float* __restrict__ w, bf16* __restrict__ o, long total)
{
  long idx = (long)blockIdx.x * blockDim.x + threadIdx.x;
  if (idx >= total) return;
  o[idx] = (bf16)w[idx];
}

__global__ void maxpool_kernel(const float* __restrict__ x, float* __restrict__ yf,
                               bf16* __restrict__ yb,
                               int N, int H, int W, int C, int OH, int OW)
{
  long total = (long)N * OH * OW * C;
  long idx = (long)blockIdx.x * blockDim.x + threadIdx.x;
  if (idx >= total) return;
  int c = (int)(idx % C); long r = idx / C;
  int ow = (int)(r % OW); r /= OW;
  int oh = (int)(r % OH); int n = (int)(r / OH);
  float m = -3.402823466e38f;
  for (int kh = 0; kh < 3; ++kh) {
    int ih = oh * 2 - 1 + kh; if (ih < 0 || ih >= H) continue;
    for (int kw = 0; kw < 3; ++kw) {
      int iw = ow * 2 - 1 + kw; if (iw < 0 || iw >= W) continue;
      float v = x[(((long)n * H + ih) * W + iw) * C + c];
      m = v > m ? v : m;
    }
  }
  yf[idx] = m; yb[idx] = (bf16)m;
}

__global__ void avgpool_kernel(const float* __restrict__ x, bf16* __restrict__ yb,
                               int N, int H, int W, int C)
{
  long idx = (long)blockIdx.x * blockDim.x + threadIdx.x;
  if (idx >= (long)N * C) return;
  int c = (int)(idx % C); int n = (int)(idx / C);
  float s = 0.f;
  for (int h = 0; h < H; ++h)
    for (int w = 0; w < W; ++w)
      s += x[(((long)n * H + h) * W + w) * C + c];
  yb[idx] = (bf16)(s / (float)(H * W));
}

__global__ void build_xin_kernel(const float* __restrict__ a_emb, int t,
                                 const float* __restrict__ em, bf16* __restrict__ xin)
{
  int idx = blockIdx.x * blockDim.x + threadIdx.x;  // 16*512
  if (idx >= 16 * 512) return;
  int b = idx / 512, j = idx % 512;
  float v = (j < 256) ? a_emb[((long)b * 64 + t) * 256 + j] : em[b * 256 + (j - 256)];
  xin[idx] = (bf16)v;
}

__global__ void lstm_point_kernel(const float* __restrict__ gates,
                                  const float* __restrict__ bih, const float* __restrict__ bhh,
                                  float* __restrict__ h, float* __restrict__ c,
                                  bf16* __restrict__ hbf, int B, int Hh)
{
  int idx = blockIdx.x * blockDim.x + threadIdx.x;
  if (idx >= B * Hh) return;
  int b = idx / Hh, j = idx % Hh;
  const float* g = gates + (long)b * 4 * Hh;
  float gi = g[j]          + bih[j]          + bhh[j];
  float gf = g[Hh + j]     + bih[Hh + j]     + bhh[Hh + j];
  float gg = g[2 * Hh + j] + bih[2 * Hh + j] + bhh[2 * Hh + j];
  float go = g[3 * Hh + j] + bih[3 * Hh + j] + bhh[3 * Hh + j];
  float si = 1.f / (1.f + expf(-gi));
  float sf = 1.f / (1.f + expf(-gf));
  float so = 1.f / (1.f + expf(-go));
  float cn = sf * c[idx] + si * tanhf(gg);
  float hn = so * tanhf(cn);
  c[idx] = cn; h[idx] = hn; hbf[idx] = (bf16)hn;
}

__global__ void outproj_kernel(const float* __restrict__ hsrc,
                               const float* __restrict__ Wout,
                               const float* __restrict__ bout,
                               float* __restrict__ out, int slot, int S1)
{
  int idx = threadIdx.x;  // 96 = 16*6
  if (idx >= 96) return;
  int b = idx / 6, o = idx % 6;
  float s = bout[o];
  for (int k = 0; k < 256; ++k) s += hsrc[b * 256 + k] * Wout[o * 256 + k];
  out[((long)b * S1 + slot) * 6 + o] = s;
}

// ---------------------------------------------------------------------------
// Host orchestration
// ---------------------------------------------------------------------------
static inline unsigned gdiv(long a, long b) { return (unsigned)((a + b - 1) / b); }
static inline int kpad32(int k) { return (k + 31) & ~31; }

struct Blk {
  const float *w1,*g1,*b1,*w2,*g2,*b2,*dw,*dg,*db;
  int has_dw, cin, ch, stride;
};
struct RNet {
  const float *stem_w,*stem_g,*stem_b;
  Blk blk[16];
  const float *fc_w,*fc_b;
};

static int parse_rnet(void* const* d_in, int cur, RNet& R) {
  auto nx = [&]() { return (const float*)d_in[cur++]; };
  static const int CH[4] = {64, 128, 256, 512};
  static const int NB[4] = {3, 4, 6, 3};
  R.stem_w = nx(); R.stem_g = nx(); R.stem_b = nx();
  int cin = 64, bi = 0;
  for (int s = 0; s < 4; ++s) {
    int ch = CH[s];
    for (int j = 0; j < NB[s]; ++j) {
      Blk& b = R.blk[bi++];
      b.w1 = nx(); b.g1 = nx(); b.b1 = nx();
      b.w2 = nx(); b.g2 = nx(); b.b2 = nx();
      b.has_dw = (j == 0 && cin != ch);
      if (b.has_dw) { b.dw = nx(); b.dg = nx(); b.db = nx(); }
      else          { b.dw = b.dg = b.db = nullptr; }
      b.cin = (j == 0) ? cin : ch;
      b.ch = ch;
      b.stride = b.has_dw ? 2 : 1;   // reference: s = 2 iff 'dw' in block
      cin = ch;
    }
  }
  R.fc_w = nx(); R.fc_b = nx();
  return cur;
}

struct WS {
  bf16*  in_bf;
  float* stem_f32;
  float *f32a, *f32b, *f32s;
  bf16  *bfa, *bfb, *bfy;
  bf16*  wt;
  bf16*  pooled;
};

static void xform_w(hipStream_t st, const float* w, bf16* wt, int O, int I, int Kk) {
  int Kp = kpad32(I * Kk * Kk);
  conv_w_transform<<<gdiv((long)O * Kp, 256), 256, 0, st>>>(w, wt, O, I, Kk, Kp);
}

static void launch_conv(hipStream_t st, const bf16* x, const bf16* wt,
                        const float* g, const float* b, const float* res,
                        float* yf, bf16* yb,
                        int N, int H, int W, int C, int OC, int OH, int OW,
                        int KH, int KW, int stride, int pad, int relu)
{
  long M = (long)N * OH * OW;
  int Kp = kpad32(KH * KW * C);
  unsigned blocks = gdiv(M, 128) * (unsigned)(OC / 64);
  if (C % 32 == 0)
    conv_wmma_kernel<true><<<blocks, 256, 0, st>>>(
        x, wt, g, b, res, yf, yb, N, H, W, C, OC, OH, OW, KH, KW, stride, pad, relu, Kp);
  else
    conv_wmma_kernel<false><<<blocks, 256, 0, st>>>(
        x, wt, g, b, res, yf, yb, N, H, W, C, OC, OH, OW, KH, KW, stride, pad, relu, Kp);
}

static void launch_gemm(hipStream_t st, const bf16* A, const bf16* Bt, float* C,
                        int M, int Nn, int K, const float* bias, int acc)
{
  long tiles = ((long)M / 16) * (Nn / 64);
  gemm_wmma_kernel<<<gdiv(tiles, 8), 256, 0, st>>>(A, Bt, C, M, Nn, K, bias, acc);
}

static void run_resnet(const RNet& R, int in_ch, const float* x_nchw,
                       int N, int H0, int W0, float* emb, const WS& ws, hipStream_t st)
{
  // input NCHW f32 -> NHWC bf16
  long tin = (long)N * in_ch * H0 * W0;
  nchw_to_nhwc_bf<<<gdiv(tin, 256), 256, 0, st>>>(x_nchw, ws.in_bf, N, in_ch, H0, W0);

  // stem 7x7 s2 p3 + BN + ReLU
  xform_w(st, R.stem_w, ws.wt, 64, in_ch, 7);
  int OH = (H0 + 6 - 7) / 2 + 1, OW = (W0 + 6 - 7) / 2 + 1;
  launch_conv(st, ws.in_bf, ws.wt, R.stem_g, R.stem_b, nullptr, ws.stem_f32, nullptr,
              N, H0, W0, in_ch, 64, OH, OW, 7, 7, 2, 3, 1);

  // 3x3 s2 p1 maxpool
  int PH = (OH - 1) / 2 + 1, PW = (OW - 1) / 2 + 1;
  maxpool_kernel<<<gdiv((long)N * PH * PW * 64, 256), 256, 0, st>>>(
      ws.stem_f32, ws.f32a, ws.bfa, N, OH, OW, 64, PH, PW);

  float* curf = ws.f32a; bf16* curb = ws.bfa;
  float* altf = ws.f32b; bf16* altb = ws.bfb;
  int H = PH, W = PW, C = 64;

  for (int bi = 0; bi < 16; ++bi) {
    const Blk& B = R.blk[bi];
    int s = B.stride;
    int oh = (H - 1) / s + 1, ow = (W - 1) / s + 1;

    const float* scp;
    if (B.has_dw) {  // downsample shortcut: 1x1 conv s2 + BN
      xform_w(st, B.dw, ws.wt, B.ch, C, 1);
      launch_conv(st, curb, ws.wt, B.dg, B.db, nullptr, ws.f32s, nullptr,
                  N, H, W, C, B.ch, oh, ow, 1, 1, s, 0, 0);
      scp = ws.f32s;
    } else {
      scp = curf;
    }
    // conv1 + BN + ReLU (bf16-only output)
    xform_w(st, B.w1, ws.wt, B.ch, C, 3);
    launch_conv(st, curb, ws.wt, B.g1, B.b1, nullptr, nullptr, ws.bfy,
                N, H, W, C, B.ch, oh, ow, 3, 3, s, 1, 1);
    // conv2 + BN + residual + ReLU (f32 + bf16 outputs)
    xform_w(st, B.w2, ws.wt, B.ch, B.ch, 3);
    launch_conv(st, ws.bfy, ws.wt, B.g2, B.b2, scp, altf, altb,
                N, oh, ow, B.ch, B.ch, oh, ow, 3, 3, 1, 1, 1);

    { float* t = curf; curf = altf; altf = t; }
    { bf16*  t = curb; curb = altb; altb = t; }
    H = oh; W = ow; C = B.ch;
  }

  // global average pool + FC 512->256 (fc_w used in natural [256,512] layout)
  avgpool_kernel<<<gdiv((long)N * C, 256), 256, 0, st>>>(curf, ws.pooled, N, H, W, C);
  cast_bf<<<gdiv(256L * 512, 256), 256, 0, st>>>(R.fc_w, ws.wt, 256L * 512);
  launch_gemm(st, ws.pooled, ws.wt, emb, N, 256, 512, R.fc_b, 0);
}

extern "C" void kernel_launch(void* const* d_in, const int* in_sizes, int n_in,
                              void* d_out, int out_size, void* d_ws, size_t ws_size,
                              hipStream_t stream) {
  (void)in_sizes; (void)n_in; (void)out_size; (void)ws_size;

  // ---- parse inputs (setup_inputs dict order, nested dicts flattened) ----
  int cur = 0;
  const float* img   = (const float*)d_in[cur++];   // [16,3,224,224]
  const float* audio = (const float*)d_in[cur++];   // [16,64,4,41]
  RNet Rimg, Raud;
  cur = parse_rnet(d_in, cur, Rimg);
  cur = parse_rnet(d_in, cur, Raud);
  const float* Wih0 = (const float*)d_in[cur++];
  const float* Whh0 = (const float*)d_in[cur++];
  const float* bih0 = (const float*)d_in[cur++];
  const float* bhh0 = (const float*)d_in[cur++];
  const float* Wih1 = (const float*)d_in[cur++];
  const float* Whh1 = (const float*)d_in[cur++];
  const float* bih1 = (const float*)d_in[cur++];
  const float* bhh1 = (const float*)d_in[cur++];
  const float* Wout = (const float*)d_in[cur++];
  const float* bout = (const float*)d_in[cur++];
  float* out = (float*)d_out;                        // [16,65,6]

  // ---- carve workspace ----
  char* base = (char*)d_ws;
  size_t off = 0;
  auto carve = [&](size_t bytes) -> void* {
    void* p = base + off;
    off += (bytes + 255) & ~(size_t)255;
    return p;
  };
  WS ws;
  ws.in_bf    = (bf16*) carve((size_t)16 * 224 * 224 * 3 * sizeof(bf16));
  ws.stem_f32 = (float*)carve((size_t)16 * 112 * 112 * 64 * sizeof(float));
  size_t actf = (size_t)16 * 56 * 56 * 64 * sizeof(float);
  size_t actb = (size_t)16 * 56 * 56 * 64 * sizeof(bf16);
  ws.f32a = (float*)carve(actf);
  ws.f32b = (float*)carve(actf);
  ws.f32s = (float*)carve(actf);
  ws.bfa  = (bf16*) carve(actb);
  ws.bfb  = (bf16*) carve(actb);
  ws.bfy  = (bf16*) carve(actb);
  ws.wt   = (bf16*) carve((size_t)512 * 512 * 9 * sizeof(bf16));
  ws.pooled = (bf16*)carve((size_t)1024 * 512 * sizeof(bf16));
  float* em0   = (float*)carve((size_t)16 * 256 * sizeof(float));
  float* a_emb = (float*)carve((size_t)1024 * 256 * sizeof(float));
  bf16* wtih0 = (bf16*)carve((size_t)1024 * 512 * sizeof(bf16));
  bf16* wthh0 = (bf16*)carve((size_t)1024 * 256 * sizeof(bf16));
  bf16* wtih1 = (bf16*)carve((size_t)1024 * 256 * sizeof(bf16));
  bf16* wthh1 = (bf16*)carve((size_t)1024 * 256 * sizeof(bf16));
  bf16*  xin   = (bf16*) carve((size_t)16 * 512 * sizeof(bf16));
  float* gates = (float*)carve((size_t)16 * 1024 * sizeof(float));
  float* h1f = (float*)carve((size_t)16 * 256 * sizeof(float));
  float* c1f = (float*)carve((size_t)16 * 256 * sizeof(float));
  float* h2f = (float*)carve((size_t)16 * 256 * sizeof(float));
  float* c2f = (float*)carve((size_t)16 * 256 * sizeof(float));
  bf16* h1b = (bf16*)carve((size_t)16 * 256 * sizeof(bf16));
  bf16* h2b = (bf16*)carve((size_t)16 * 256 * sizeof(bf16));

  // ---- two ResNet34 towers ----
  run_resnet(Rimg, 3, img,   16,   224, 224, em0,   ws, stream);  // em0  [16,256]
  run_resnet(Raud, 1, audio, 1024, 4,   41,  a_emb, ws, stream);  // a_emb[16*64,256]

  // ---- LSTM weights: cast to bf16 (natural [4H,K] layout, no transpose) ----
  cast_bf<<<gdiv(1024L * 512, 256), 256, 0, stream>>>(Wih0, wtih0, 1024L * 512);
  cast_bf<<<gdiv(1024L * 256, 256), 256, 0, stream>>>(Whh0, wthh0, 1024L * 256);
  cast_bf<<<gdiv(1024L * 256, 256), 256, 0, stream>>>(Wih1, wtih1, 1024L * 256);
  cast_bf<<<gdiv(1024L * 256, 256), 256, 0, stream>>>(Whh1, wthh1, 1024L * 256);

  // ---- zero LSTM state ----
  (void)hipMemsetAsync(h1f, 0, (size_t)16 * 256 * sizeof(float), stream);
  (void)hipMemsetAsync(c1f, 0, (size_t)16 * 256 * sizeof(float), stream);
  (void)hipMemsetAsync(h2f, 0, (size_t)16 * 256 * sizeof(float), stream);
  (void)hipMemsetAsync(c2f, 0, (size_t)16 * 256 * sizeof(float), stream);
  (void)hipMemsetAsync(h1b, 0, (size_t)16 * 256 * sizeof(bf16), stream);
  (void)hipMemsetAsync(h2b, 0, (size_t)16 * 256 * sizeof(bf16), stream);

  // ---- pose from initial image embedding ----
  outproj_kernel<<<1, 96, 0, stream>>>(em0, Wout, bout, out, 0, 65);

  // ---- 64 sequential LSTM steps ----
  for (int t = 0; t < 64; ++t) {
    const float* em = (t == 0) ? em0 : h2f;
    build_xin_kernel<<<gdiv(16 * 512, 256), 256, 0, stream>>>(a_emb, t, em, xin);
    // layer 1 gates = xin @ Wih0^T + h1 @ Whh0^T   (biases in pointwise)
    launch_gemm(stream, xin, wtih0, gates, 16, 1024, 512, nullptr, 0);
    launch_gemm(stream, h1b, wthh0, gates, 16, 1024, 256, nullptr, 1);
    lstm_point_kernel<<<gdiv(16 * 256, 256), 256, 0, stream>>>(
        gates, bih0, bhh0, h1f, c1f, h1b, 16, 256);
    // layer 2
    launch_gemm(stream, h1b, wtih1, gates, 16, 1024, 256, nullptr, 0);
    launch_gemm(stream, h2b, wthh1, gates, 16, 1024, 256, nullptr, 1);
    lstm_point_kernel<<<gdiv(16 * 256, 256), 256, 0, stream>>>(
        gates, bih1, bhh1, h2f, c2f, h2b, 16, 256);
    // pose output for step t -> slot t+1
    outproj_kernel<<<1, 96, 0, stream>>>(h2f, Wout, bout, out, t + 1, 65);
  }
}
